// LinearAttention_54778012893581
// MI455X (gfx1250) — matrix operand order repaired and checked by reading, not compiled, not run on hardware
//
#include <hip/hip_runtime.h>
#include <hip/hip_bf16.h>

// ---------------------------------------------------------------------------
// LinearAttention for MI455X (gfx1250, wave32, WMMA + TDM).
// All heavy GEMMs use v_wmma_f32_16x16x32_f16 (f32 accumulate, f16 operands).
// Memory-bound workload (~37 GFLOP vs ~0.5GB traffic @ 23.3TB/s) -> f16
// intermediates, fragment loads as contiguous 16B vectors, and the context
// kernel stages its LDS tiles with the Tensor Data Mover (tensor_load_to_lds,
// TENSORcnt) instead of VGPR round-trips.
// ---------------------------------------------------------------------------

typedef __attribute__((ext_vector_type(16))) _Float16 v16h;
typedef __attribute__((ext_vector_type(8)))  _Float16 v8h;
typedef __attribute__((ext_vector_type(8)))  float    v8f;
typedef __attribute__((ext_vector_type(4)))  int      v4i;
typedef __attribute__((ext_vector_type(4)))  unsigned v4u;
typedef __attribute__((ext_vector_type(8)))  int      v8i;

#define DIMC   256
#define INNER  256
#define HEADS  8
#define DH     32
#define NPIX   4096          // 64*64 per image
#define BATCH  16
#define MTOT   (BATCH*NPIX)  // 65536 pixel rows
#define SCALE  0.17677669529663687f   // 32^-0.5
#define EPS    1e-5f

union Frag16 { v16h v; v8h h[2]; };

// ---- WMMA fragment helpers (wave32; A/B use contiguous k order on both operands, not the ISA table order) ----
// A 16x32 f16: lane<16 -> row=lane, K = 0..15
//              lane>=16 -> row=lane-16, K = 16..31
__device__ __forceinline__ v16h load_a_frag(const _Float16* A, int lda,
                                            long m0, int k0) {
  int lane = threadIdx.x & 31;
  int r    = lane & 15;
  int hi   = lane >> 4;
  const _Float16* p = A + (long)(m0 + r) * lda + k0 + hi * 16;
  Frag16 f;
  f.h[0] = *(const v8h*)(p);
  f.h[1] = *(const v8h*)(p + 8);
  return f.v;
}

// B 32x16 f16 from row-major Wt[n][k] (columns of B are contiguous K runs):
// lane<16 -> col=lane, halves K = 0..15 ; lane>=16 -> col=lane-16, K = 16..31
__device__ __forceinline__ v16h load_b_frag(const _Float16* Wt, int ldb,
                                            long n0, int k0) {
  int lane = threadIdx.x & 31;
  int c    = lane & 15;
  int hi   = lane >> 4;
  const _Float16* p = Wt + (long)(n0 + c) * ldb + k0 + hi * 16;
  Frag16 f;
  f.h[0] = *(const v8h*)(p);
  f.h[1] = *(const v8h*)(p + 8);
  return f.v;
}

// C/D 16x16 f32: VGPR r -> (M = r + (lane>=16 ? 8 : 0), N = lane&15)
__device__ __forceinline__ void store_d_f16(_Float16* C, int ldc,
                                            long m0, long n0, v8f acc) {
  int lane = threadIdx.x & 31;
  int col  = lane & 15;
  int rb   = (lane >> 4) * 8;
#pragma unroll
  for (int r = 0; r < 8; ++r)
    C[(long)(m0 + rb + r) * ldc + n0 + col] = (_Float16)acc[r];
}

__device__ __forceinline__ void store_d_f32(float* C, int ldc,
                                            long m0, long n0, v8f acc) {
  int lane = threadIdx.x & 31;
  int col  = lane & 15;
  int rb   = (lane >> 4) * 8;
#pragma unroll
  for (int r = 0; r < 8; ++r)
    C[(long)(m0 + rb + r) * ldc + n0 + col] = acc[r];
}

// ---------------------------------------------------------------------------
// TDM: wave-issued 2-D tile DMA global -> LDS (CDNA5 ISA).
// 16-bit elements; tile = tile_d0 x tile_d1, row stride = stride0 elements.
// Destination LDS layout: rows packed back-to-back (tile_d0 elems per row).
// This toolchain exposes the 6-arg builtin:
//   (uint32x4 g0, int32x8 g1, int32x4 g2, int32x4 g3, int32x8 gx, i32 cpol)
// Our tensor is 2-D, so groups 2/3 (and the extra payload) are zero.
// ---------------------------------------------------------------------------
__device__ __forceinline__ void tdm_load_2d_f16(unsigned lds_off,
                                                unsigned long long gaddr,
                                                unsigned tile_d0, unsigned tile_d1,
                                                unsigned long long stride0) {
  v4u g0;
  g0[0] = 1u;                                           // count=1, user D#
  g0[1] = lds_off;                                      // lds_addr (bytes)
  g0[2] = (unsigned)(gaddr & 0xffffffffull);            // global_addr[31:0]
  g0[3] = (unsigned)((gaddr >> 32) & 0x1ffffffull)      // global_addr[56:32]
        | (2u << 30);                                   // type = 2 ("image")
  v8i g1;
  g1[0] = 0x00010000;                                   // data_size=1 (2B)
  g1[1] = (int)((tile_d0 & 0xffffu) << 16);             // tensor_dim0 lo16
  g1[2] = (int)((tile_d0 >> 16) | ((tile_d1 & 0xffffu) << 16)); // dim0 hi/dim1 lo
  g1[3] = (int)((tile_d1 >> 16) | (tile_d0 << 16));     // dim1 hi | tile_dim0
  g1[4] = (int)(tile_d1 & 0xffffu);                     // tile_dim1 (tile_dim2=0)
  g1[5] = (int)(stride0 & 0xffffffffull);               // tensor_dim0_stride lo
  g1[6] = (int)((stride0 >> 32) & 0xffffull);           // stride hi (dim1_stride=0)
  g1[7] = 0;
  v4i gz = {0, 0, 0, 0};                                // groups 2/3: 2-D tensor
  v8i gx = {0, 0, 0, 0, 0, 0, 0, 0};                    // extra payload: unused
  __builtin_amdgcn_tensor_load_to_lds(g0, g1, gz, gz, gx, 0);
}

// ---------------------------------------------------------------------------
// Kernel 0: cast weights to f16 (B layout is already [o][c], K-contiguous),
// zero the k-softmax denominators (re-done each launch: graph-replay safe).
// ---------------------------------------------------------------------------
__global__ __launch_bounds__(256)
void prep_kernel(const float* __restrict__ Wq1, const float* __restrict__ Wk1,
                 const float* __restrict__ Wv1, const float* __restrict__ Wo,
                 _Float16* __restrict__ wqkv, _Float16* __restrict__ wo,
                 float* __restrict__ ksum) {
  int i = blockIdx.x * 256 + threadIdx.x;            // grid 256 -> i < 65536
  wqkv[i]          = (_Float16)Wq1[i];
  wqkv[65536 + i]  = (_Float16)Wk1[i];
  wqkv[131072 + i] = (_Float16)Wv1[i];
  wo[i]            = (_Float16)Wo[i];
  if (i < BATCH * DIMC) ksum[i] = 0.f;
}

// ---------------------------------------------------------------------------
// Kernel 1: channel LayerNorm of fmap (NCHW) -> xln[pixel][256] f16.
// Block = 256 thr = 64 pixels; thread (tx,ty) owns channels ty*64..+63.
// fmap read twice (2nd read hits L2: 64MB < 192MB). LDS transpose gives
// fully coalesced 128B-per-thread f16 row writes.
// ---------------------------------------------------------------------------
__global__ __launch_bounds__(256)
void ln_in_kernel(const float* __restrict__ fmap, const float* __restrict__ g,
                  _Float16* __restrict__ xln) {
  __shared__ float redS[4][64];
  __shared__ float redQ[4][64];
  __shared__ float meanv[64], rstdv[64];
  __shared__ __align__(16) _Float16 tile[64 * 256];

  int b   = blockIdx.x >> 6;
  int xy0 = (blockIdx.x & 63) * 64;
  int tx  = threadIdx.x & 63;
  int ty  = threadIdx.x >> 6;

  const float* base = fmap + ((long)b * DIMC + ty * 64) * NPIX + xy0 + tx;
  float s = 0.f, q = 0.f;
#pragma unroll 8
  for (int j = 0; j < 64; ++j) {
    float v = base[(long)j * NPIX];
    s += v; q += v * v;
  }
  redS[ty][tx] = s; redQ[ty][tx] = q;
  __syncthreads();
  if (ty == 0) {
    float S = redS[0][tx] + redS[1][tx] + redS[2][tx] + redS[3][tx];
    float Q = redQ[0][tx] + redQ[1][tx] + redQ[2][tx] + redQ[3][tx];
    float mean = S * (1.f / 256.f);
    float var  = Q * (1.f / 256.f) - mean * mean;
    meanv[tx] = mean;
    rstdv[tx] = rsqrtf(var + EPS);
  }
  __syncthreads();
  float mean = meanv[tx], rstd = rstdv[tx];
#pragma unroll 8
  for (int j = 0; j < 64; ++j) {
    int c = ty * 64 + j;
    float v = base[(long)j * NPIX];
    tile[tx * 256 + c] = (_Float16)((v - mean) * rstd * g[c]);
  }
  __syncthreads();
  int p   = threadIdx.x >> 2;
  int ch0 = (threadIdx.x & 3) * 64;
  const v4i* src = (const v4i*)&tile[p * 256 + ch0];
  v4i* dst = (v4i*)(xln + ((long)(b * NPIX + xy0 + p)) * 256 + ch0);
#pragma unroll
  for (int j = 0; j < 8; ++j) dst[j] = src[j];
}

// ---------------------------------------------------------------------------
// Kernel 2: GEMM  out[M][N] = A[M][256] * Wt[N][256]^T  (K = 256 fixed).
// Wave tile 16Mx64N (4 accumulators), block = 8 waves = 128M x 64N.
// Used for QKV projection (N=768, f16 out) and out-proj (N=256, f32 out).
// ---------------------------------------------------------------------------
__global__ __launch_bounds__(256)
void gemm_wmma_k256(const _Float16* __restrict__ A,
                    const _Float16* __restrict__ Wt,
                    _Float16* __restrict__ outh, float* __restrict__ outf,
                    int N) {
  int wave = threadIdx.x >> 5;
  long m0 = (long)blockIdx.x * 128 + wave * 16;
  long n0 = (long)blockIdx.y * 64;

  v8f acc0 = {}, acc1 = {}, acc2 = {}, acc3 = {};
#pragma unroll
  for (int k0 = 0; k0 < 256; k0 += 32) {
    v16h a  = load_a_frag(A, 256, m0, k0);
    v16h b0 = load_b_frag(Wt, 256, n0,      k0);
    v16h b1 = load_b_frag(Wt, 256, n0 + 16, k0);
    v16h b2 = load_b_frag(Wt, 256, n0 + 32, k0);
    v16h b3 = load_b_frag(Wt, 256, n0 + 48, k0);
    acc0 = __builtin_amdgcn_wmma_f32_16x16x32_f16(false, a, false, b0, (short)0, acc0, false, false);
    acc1 = __builtin_amdgcn_wmma_f32_16x16x32_f16(false, a, false, b1, (short)0, acc1, false, false);
    acc2 = __builtin_amdgcn_wmma_f32_16x16x32_f16(false, a, false, b2, (short)0, acc2, false, false);
    acc3 = __builtin_amdgcn_wmma_f32_16x16x32_f16(false, a, false, b3, (short)0, acc3, false, false);
  }
  if (outh) {
    store_d_f16(outh, N, m0, n0,      acc0);
    store_d_f16(outh, N, m0, n0 + 16, acc1);
    store_d_f16(outh, N, m0, n0 + 32, acc2);
    store_d_f16(outh, N, m0, n0 + 48, acc3);
  } else {
    store_d_f32(outf, N, m0, n0,      acc0);
    store_d_f32(outf, N, m0, n0 + 16, acc1);
    store_d_f32(outf, N, m0, n0 + 32, acc2);
    store_d_f32(outf, N, m0, n0 + 48, acc3);
  }
}

// ---------------------------------------------------------------------------
// Kernel 3: depthwise 3x3 conv on t[pixel][768] + per-mode epilogue.
// One thread = one (pixel, head): computes 32 channels (contiguous 64B rows).
//   mode 0 (q): softmax over the 32 head channels, * SCALE
//   mode 1 (k): exp(u); per-(b,channel) sums via LDS + one global atomic
//   mode 2 (v): plain
// ---------------------------------------------------------------------------
__global__ __launch_bounds__(256)
void dwconv_kernel(const _Float16* __restrict__ t,   // [65536][768]
                   const float* __restrict__ wdw,    // [256][9]
                   _Float16* __restrict__ outp,      // [65536][256]
                   float* __restrict__ ksum,         // [16][256] (mode 1)
                   int chanBase, int mode) {
  __shared__ float partial[256];
  if (mode == 1) partial[threadIdx.x] = 0.f;
  __syncthreads();

  int b  = blockIdx.x >> 7;                 // 128 blocks / image
  int i  = (blockIdx.x & 127) * 32 + (threadIdx.x >> 3);
  int h  = threadIdx.x & 7;
  int x  = i >> 6, y = i & 63;

  float u[32];
#pragma unroll
  for (int c = 0; c < 32; ++c) u[c] = 0.f;

#pragma unroll
  for (int di = 0; di < 3; ++di) {
#pragma unroll
    for (int dj = 0; dj < 3; ++dj) {
      int xx = x + di - 1, yy = y + dj - 1;
      if (xx < 0 || xx > 63 || yy < 0 || yy > 63) continue;
      const _Float16* row =
          t + ((long)b * NPIX + xx * 64 + yy) * 768 + chanBase + h * DH;
      const float* w = wdw + (h * DH) * 9 + di * 3 + dj;
#pragma unroll
      for (int c = 0; c < 32; ++c)
        u[c] += w[c * 9] * (float)row[c];
    }
  }

  _Float16* dst = outp + ((long)b * NPIX + i) * 256 + h * DH;
  if (mode == 0) {                       // q: softmax over feature dim
    float m = u[0];
#pragma unroll
    for (int c = 1; c < 32; ++c) m = fmaxf(m, u[c]);
    float s = 0.f;
#pragma unroll
    for (int c = 0; c < 32; ++c) { u[c] = __expf(u[c] - m); s += u[c]; }
    float inv = SCALE / s;
#pragma unroll
    for (int c = 0; c < 32; ++c) dst[c] = (_Float16)(u[c] * inv);
  } else if (mode == 1) {                // k: exp + per-channel sums
#pragma unroll
    for (int c = 0; c < 32; ++c) {
      float e = __expf(u[c]);
      dst[c] = (_Float16)e;
      atomicAdd(&partial[h * DH + c], e);
    }
  } else {                               // v: plain
#pragma unroll
    for (int c = 0; c < 32; ++c) dst[c] = (_Float16)u[c];
  }
  __syncthreads();
  if (mode == 1)
    atomicAdd(&ksum[b * DIMC + threadIdx.x], partial[threadIdx.x]);
}

// ---------------------------------------------------------------------------
// Kernel 4: per-(b,h) context = expk^T (32x4096) x v (4096x32), K split
// across 8 waves. Tiles staged into LDS by the Tensor Data Mover (one
// 256x32-element 2-D descriptor per tensor per step, issued by wave 0,
// synced with s_wait_tensorcnt + block barrier). Folds 1/ksum[d]; stores
// transposed ctxT[bh][e][d] f16 so kernel 5's B fragments are contiguous.
// ---------------------------------------------------------------------------
__global__ __launch_bounds__(256)
void context_kernel(const _Float16* __restrict__ ek,
                    const _Float16* __restrict__ vv,
                    const float* __restrict__ ksum,
                    _Float16* __restrict__ ctxT) {
  __shared__ __align__(16) char smem[32768];
  _Float16* sE = (_Float16*)smem;            // [256][32]  16KB (TDM dest)
  _Float16* sV = sE + 256 * 32;              // [256][32]  16KB (TDM dest)
  // after the K loop the same 32KB is reused as float sAcc[8][32][32]

  int bh = blockIdx.x, b = bh >> 3, h = bh & 7;
  int wave = threadIdx.x >> 5, lane = threadIdx.x & 31;
  int kw = wave * 32;

  unsigned sE_off = (unsigned)(unsigned long long)(void*)sE;   // LDS byte addr
  unsigned sV_off = sE_off + 16384;

  v8f acc00 = {}, acc01 = {}, acc10 = {}, acc11 = {};

  for (int n0 = 0; n0 < NPIX; n0 += 256) {
    __syncthreads();                       // previous tiles fully consumed
    if (wave == 0) {
      unsigned long long ge = (unsigned long long)(void*)
          (ek + ((long)b * NPIX + n0) * 256 + h * DH);
      unsigned long long gv = (unsigned long long)(void*)
          (vv + ((long)b * NPIX + n0) * 256 + h * DH);
      tdm_load_2d_f16(sE_off, ge, 32, 256, 256);   // 256 rows x 32 halves
      tdm_load_2d_f16(sV_off, gv, 32, 256, 256);
      __builtin_amdgcn_s_wait_tensorcnt(0);
    }
    __syncthreads();

    // A fragments (rows = d), gathered from sE
    Frag16 a0, a1;
    {
      int r = lane & 15, hi = lane >> 4;
#pragma unroll
      for (int j = 0; j < 8; ++j) {
        int k0 = hi * 16 + j, k1 = hi * 16 + 8 + j;
        a0.v[j]     = sE[(kw + k0) * 32 + r];
        a0.v[j + 8] = sE[(kw + k1) * 32 + r];
        a1.v[j]     = sE[(kw + k0) * 32 + 16 + r];
        a1.v[j + 8] = sE[(kw + k1) * 32 + 16 + r];
      }
    }
    // B fragments (cols = e), gathered from sV
    Frag16 b0, b1;
    {
      int c = lane & 15, hi = lane >> 4;
#pragma unroll
      for (int j = 0; j < 16; ++j) {
        b0.v[j] = sV[(kw + hi * 16 + j) * 32 + c];
        b1.v[j] = sV[(kw + hi * 16 + j) * 32 + 16 + c];
      }
    }
    acc00 = __builtin_amdgcn_wmma_f32_16x16x32_f16(false, a0.v, false, b0.v, (short)0, acc00, false, false);
    acc01 = __builtin_amdgcn_wmma_f32_16x16x32_f16(false, a0.v, false, b1.v, (short)0, acc01, false, false);
    acc10 = __builtin_amdgcn_wmma_f32_16x16x32_f16(false, a1.v, false, b0.v, (short)0, acc10, false, false);
    acc11 = __builtin_amdgcn_wmma_f32_16x16x32_f16(false, a1.v, false, b1.v, (short)0, acc11, false, false);
  }

  __syncthreads();
  float* sAcc = (float*)smem;                // [8][32][32]
  {
    int col = lane & 15, rb = (lane >> 4) * 8;
    float* w = sAcc + wave * 1024;
#pragma unroll
    for (int r = 0; r < 8; ++r) {
      w[(rb + r) * 32 + col]            = acc00[r];
      w[(rb + r) * 32 + 16 + col]       = acc01[r];
      w[(16 + rb + r) * 32 + col]       = acc10[r];
      w[(16 + rb + r) * 32 + 16 + col]  = acc11[r];
    }
  }
  __syncthreads();
  for (int t = threadIdx.x; t < 1024; t += 256) {
    int d = t >> 5, e = t & 31;
    float s = 0.f;
#pragma unroll
    for (int w = 0; w < 8; ++w) s += sAcc[w * 1024 + d * 32 + e];
    float denom = ksum[b * DIMC + h * DH + d];
    ctxT[(long)bh * 1024 + e * 32 + d] = (_Float16)(s / denom);
  }
}

// ---------------------------------------------------------------------------
// Kernel 5: out = q (Mx32) x context (32x32), SiLU, f16 store. One wave per
// (bh, 16-pixel tile): K=32 -> a single WMMA per output tile.
// ---------------------------------------------------------------------------
__global__ __launch_bounds__(256)
void attn_out_kernel(const _Float16* __restrict__ q,     // [65536][256]
                     const _Float16* __restrict__ ctxT,  // [128][32][32]
                     _Float16* __restrict__ attn) {      // [65536][256]
  int wid = blockIdx.x * 8 + (threadIdx.x >> 5);
  int bh = wid >> 8, pt = wid & 255;
  int b = bh >> 3, h = bh & 7;
  long m0 = (long)b * NPIX + pt * 16;

  // A fragment straight from global (rows contiguous in K = head channels)
  int lane = threadIdx.x & 31;
  int r = lane & 15, hi = lane >> 4;
  Frag16 a;
  {
    const _Float16* p = q + (m0 + r) * 256 + h * DH + hi * 16;
    a.h[0] = *(const v8h*)(p);
    a.h[1] = *(const v8h*)(p + 8);
  }
  Frag16 b0, b1;
  {
    const _Float16* p0 = ctxT + (long)bh * 1024 + (long)(lane & 15) * 32 + hi * 16;
    b0.h[0] = *(const v8h*)(p0);
    b0.h[1] = *(const v8h*)(p0 + 8);
    const _Float16* p1 = p0 + 16 * 32;
    b1.h[0] = *(const v8h*)(p1);
    b1.h[1] = *(const v8h*)(p1 + 8);
  }
  v8f z = {};
  v8f acc0 = __builtin_amdgcn_wmma_f32_16x16x32_f16(false, a.v, false, b0.v, (short)0, z, false, false);
  v8f acc1 = __builtin_amdgcn_wmma_f32_16x16x32_f16(false, a.v, false, b1.v, (short)0, z, false, false);

  int col = lane & 15, rb = (lane >> 4) * 8;
#pragma unroll
  for (int rr = 0; rr < 8; ++rr) {
    float x0 = acc0[rr], x1 = acc1[rr];
    x0 = x0 / (1.f + __expf(-x0));          // SiLU
    x1 = x1 / (1.f + __expf(-x1));
    _Float16* d = attn + (m0 + rb + rr) * 256 + h * DH;
    d[col]      = (_Float16)x0;
    d[16 + col] = (_Float16)x1;
  }
}

// ---------------------------------------------------------------------------
// Kernel 7: final channel LayerNorm. y[pixel][256] f32 (row-contiguous,
// coalesced) -> d_out NCHW f32 (coalesced across the 64-pixel tx axis).
// ---------------------------------------------------------------------------
__global__ __launch_bounds__(256)
void ln_out_kernel(const float* __restrict__ y, const float* __restrict__ g,
                   float* __restrict__ out) {
  __shared__ float redS[4][64];
  __shared__ float redQ[4][64];
  __shared__ float meanv[64], rstdv[64];

  int b   = blockIdx.x >> 6;
  int xy0 = (blockIdx.x & 63) * 64;
  int tx  = threadIdx.x & 63;
  int ty  = threadIdx.x >> 6;

  const float* row = y + ((long)(b * NPIX + xy0 + tx)) * 256 + ty * 64;
  float s = 0.f, q = 0.f;
#pragma unroll 8
  for (int j = 0; j < 64; ++j) { float v = row[j]; s += v; q += v * v; }
  redS[ty][tx] = s; redQ[ty][tx] = q;
  __syncthreads();
  if (ty == 0) {
    float S = redS[0][tx] + redS[1][tx] + redS[2][tx] + redS[3][tx];
    float Q = redQ[0][tx] + redQ[1][tx] + redQ[2][tx] + redQ[3][tx];
    float mean = S * (1.f / 256.f);
    float var  = Q * (1.f / 256.f) - mean * mean;
    meanv[tx] = mean;
    rstdv[tx] = rsqrtf(var + EPS);
  }
  __syncthreads();
  float mean = meanv[tx], rstd = rstdv[tx];
#pragma unroll 8
  for (int j = 0; j < 64; ++j) {
    int c = ty * 64 + j;
    out[((long)b * DIMC + c) * NPIX + xy0 + tx] = (row[j] - mean) * rstd * g[c];
  }
}

// ---------------------------------------------------------------------------
// Host orchestration. Workspace layout (bytes):
//   [0, 32M)        xln f16         (reused for q after QKV GEMM)
//   [32M, 128M)     t[65536][768]   (reused: attn f16 @32M, y f32 @64M)
//   [128M, 160M)    expk f16
//   [160M, 192M)    v f16
//   [192M ...)      ksum(16KB) | ctxT(256KB) | wqkv f16(384KB) | wo f16(128KB)
// ---------------------------------------------------------------------------
extern "C" void kernel_launch(void* const* d_in, const int* in_sizes, int n_in,
                              void* d_out, int out_size, void* d_ws, size_t ws_size,
                              hipStream_t stream) {
  const float* fmap  = (const float*)d_in[0];
  const float* g_in  = (const float*)d_in[1];
  const float* Wq1   = (const float*)d_in[2];
  const float* Wq_dw = (const float*)d_in[3];
  const float* Wk1   = (const float*)d_in[4];
  const float* Wk_dw = (const float*)d_in[5];
  const float* Wv1   = (const float*)d_in[6];
  const float* Wv_dw = (const float*)d_in[7];
  const float* Wo    = (const float*)d_in[8];
  const float* g_out = (const float*)d_in[9];
  float* outp = (float*)d_out;

  char* ws = (char*)d_ws;
  const size_t MB = 1024ull * 1024ull;
  _Float16* xln  = (_Float16*)(ws);                    // 32 MB (later: q)
  _Float16* tbuf = (_Float16*)(ws + 32 * MB);          // 96 MB
  _Float16* ekb  = (_Float16*)(ws + 128 * MB);         // 32 MB
  _Float16* vbuf = (_Float16*)(ws + 160 * MB);         // 32 MB
  float*    ksum = (float*)   (ws + 192 * MB);         // 16 KB
  _Float16* ctxT = (_Float16*)(ws + 192 * MB + 64 * 1024);   // 256 KB
  _Float16* wqkv = (_Float16*)(ws + 192 * MB + 512 * 1024);  // 384 KB
  _Float16* wo_h = (_Float16*)(ws + 193 * MB);               // 128 KB
  _Float16* qbuf = xln;                                // reuse after GEMM1
  _Float16* attn = tbuf;                               // reuse after dwconv
  float*    ybuf = (float*)(ws + 64 * MB);             // reuse tbuf tail

  // 0) weights -> f16, zero ksum (every call: graph-replay deterministic)
  prep_kernel<<<256, 256, 0, stream>>>(Wq1, Wk1, Wv1, Wo, wqkv, wo_h, ksum);

  // 1) channel LN -> xln[pixel][256]
  ln_in_kernel<<<1024, 256, 0, stream>>>(fmap, g_in, xln);

  // 2) fused QKV projection: [65536,256] x [256,768] (WMMA)
  gemm_wmma_k256<<<dim3(512, 12), 256, 0, stream>>>(xln, wqkv, tbuf, nullptr, 768);

  // 3) depthwise 3x3 + softmax epilogues
  dwconv_kernel<<<2048, 256, 0, stream>>>(tbuf, Wq_dw, qbuf, ksum, 0,   0);
  dwconv_kernel<<<2048, 256, 0, stream>>>(tbuf, Wk_dw, ekb,  ksum, 256, 1);
  dwconv_kernel<<<2048, 256, 0, stream>>>(tbuf, Wv_dw, vbuf, ksum, 512, 2);

  // 4) per-head context (WMMA + TDM-staged LDS tiles, folds 1/ksum)
  context_kernel<<<128, 256, 0, stream>>>(ekb, vbuf, ksum, ctxT);

  // 5) out = q x context, SiLU (WMMA)
  attn_out_kernel<<<4096, 256, 0, stream>>>(qbuf, ctxT, attn);

  // 6) output projection: [65536,256] x [256,256] -> f32 (WMMA)
  gemm_wmma_k256<<<dim3(512, 4), 256, 0, stream>>>(attn, wo_h, nullptr, ybuf, 256);

  // 7) final channel LN -> NCHW f32 output
  ln_out_kernel<<<1024, 256, 0, stream>>>(ybuf, g_out, outp);
}
